// SimplifiedGateNetwork_23957327577730
// MI455X (gfx1250) — matrix operand order repaired and checked by reading, not compiled, not run on hardware
//
#include <hip/hip_runtime.h>
#include <hip/hip_bf16.h>

// ---------------------------------------------------------------------------
// Bidirectional-LSTM gate network for MI455X (gfx1250, wave32, WMMA).
//
// x_t @ Wih.T depends only on the token id (VOCAB=40) -> precompute
// xW[d][tok][128] = Wih_d @ emb[tok] + bih_d + bhh_d (kernel A); the input
// projection becomes an LDS gather that is fed into the WMMA C operand.
// Recurrence h(16x32) @ Whh.T(32x128) = 8x V_WMMA_F32_16X16X32_F16 per
// wave-step (M=16 sequences, N=16 gates, K=HID=32).  Activations use the
// gfx1250 hardware V_TANH_F32 (sigmoid(x) = 0.5*tanh(x/2)+0.5), branch-free.
// ---------------------------------------------------------------------------

typedef __attribute__((ext_vector_type(16))) _Float16 v16h;
typedef __attribute__((ext_vector_type(8)))  _Float16 v8h;
typedef __attribute__((ext_vector_type(8)))  float    v8f;

#define VOCAB 40
#define DM    128
#define HID   32
#define G4    128   // 4*HID gates
#define SEQ   40
#define BATCH 16384
#define WPB   4     // waves per block
#define SPW   16    // sequences per wave

// workspace layout
#define WS_XW_FLOATS (2 * VOCAB * G4)        // 10240 floats (fwd+bwd xW tables)
#define WS_WHH_OFF   (WS_XW_FLOATS * 4)      // byte offset 40960

__device__ __forceinline__ float tanh_approx(float x) {
#if defined(__HIP_DEVICE_COMPILE__)
# if __has_builtin(__builtin_amdgcn_tanhf)
  return __builtin_amdgcn_tanhf(x);             // gfx1250 V_TANH_F32
# elif __has_builtin(__builtin_amdgcn_tanh_f32)
  return __builtin_amdgcn_tanh_f32(x);
# else
  const float ax = __builtin_fabsf(x);          // branch-free fallback
  const float e  = __expf(-2.0f * ax);
  return __builtin_copysignf((1.0f - e) / (1.0f + e), x);
# endif
#else
  return tanhf(x);
#endif
}
__device__ __forceinline__ float sigmoid_t(float x) {
  return __builtin_fmaf(tanh_approx(0.5f * x), 0.5f, 0.5f);
}

// ---------------------------------------------------------------------------
// Kernel A: build xW tables (f32) and Whh in WMMA B-matrix register layout
// (f16).  For a 32x16 f16 B tile: N = lane & 15, element j holds K = 8*(lane >> 4) + j + (j & 8),
// B[k][n] = Whh[tile*16+n][k].
// ---------------------------------------------------------------------------
__global__ void precompute_kernel(const float* __restrict__ emb,
                                  const float* __restrict__ WihF, const float* __restrict__ WhhF,
                                  const float* __restrict__ bihF, const float* __restrict__ bhhF,
                                  const float* __restrict__ WihB, const float* __restrict__ WhhB,
                                  const float* __restrict__ bihB, const float* __restrict__ bhhB,
                                  float* __restrict__ ws_xw, _Float16* __restrict__ ws_whh)
{
  const int b = blockIdx.x, tid = threadIdx.x;
  if (b < 2 * VOCAB) {
    const int d = b / VOCAB, t = b % VOCAB;
    const float* Wih = d ? WihB : WihF;
    const float* bi  = d ? bihB : bihF;
    const float* bh  = d ? bhhB : bhhF;
    const int g = tid;                       // gate 0..127
    float s = bi[g] + bh[g];
    const float* e = emb + (size_t)t * DM;
    for (int k = 0; k < DM; ++k) s += e[k] * Wih[g * DM + k];
    ws_xw[(d * VOCAB + t) * G4 + g] = s;
  } else {
    const int chunk = b - 2 * VOCAB;         // 0..15
    const int base = chunk * 512 + tid * 4;
    #pragma unroll
    for (int q = 0; q < 4; ++q) {
      const int e = base + q;                // 0..8191
      const int d    = e >> 12;
      const int rem  = e & 4095;
      const int t    = rem >> 9;             // gate tile 0..7
      const int r2   = rem & 511;
      const int lane = r2 >> 4;              // N = lane & 15, half h = lane >> 4
      const int j    = r2 & 15;              // element: K = 8h + j + (j & 8)
      const float* Whh = d ? WhhB : WhhF;    // shape (128, 32) row-major
      ws_whh[e] = (_Float16)Whh[(t * 16 + (lane & 15)) * HID + 8 * (lane >> 4) + j + (j & 8)];
    }
  }
}

// ---------------------------------------------------------------------------
// Kernel B: full network. One wave = 16 sequences.
// ---------------------------------------------------------------------------
__global__ void __launch_bounds__(WPB * 32)
lstm_gate_kernel(const int* __restrict__ x,
                 const float* __restrict__ ws_xw, const _Float16* __restrict__ ws_whh,
                 const float* __restrict__ W1, const float* __restrict__ b1,
                 const float* __restrict__ W2, const float* __restrict__ b2,
                 const float* __restrict__ ew, const float* __restrict__ lw,
                 float* __restrict__ out)
{
  __shared__ float          s_xW[VOCAB * G4];          // current direction's table (20 KB)
  __shared__ float          s_W1[32 * 66];
  __shared__ float          s_b1[32];
  __shared__ float          s_W2[4 * 32];
  __shared__ float          s_b2[4];
  __shared__ float          s_scal[2];
  __shared__ unsigned short s_tok[WPB][SPW][SEQ];
  __shared__ _Float16       s_hT[WPB][SPW][HID];       // h transpose buffer (f16)
  __shared__ float          s_hf[WPB][SPW][HID];       // final forward h
  __shared__ float          s_hb[WPB][SPW][HID];       // final backward h

  const int tid   = threadIdx.x;
  const int wave  = tid >> 5;
  const int lane  = tid & 31;
  const int group = lane >> 4;    // C/D halves: lanes 0-15 -> rows 0-7, 16-31 -> rows 8-15
  const int nn    = lane & 15;    // column within 16-wide gate tile / A-matrix row
  const int seq0  = (blockIdx.x * WPB + wave) * SPW;

  // ---- cooperative block loads -------------------------------------------
  for (int i = tid; i < 32 * 66; i += WPB * 32) s_W1[i] = W1[i];
  if (tid < 32) s_b1[tid] = b1[tid];
  if (tid < 4 * 32) s_W2[tid] = W2[tid];
  if (tid < 4) s_b2[tid] = b2[tid];
  if (tid == 0) { s_scal[0] = ew[0]; s_scal[1] = lw[0]; }
  for (int i = lane; i < SPW * SEQ; i += 32) {
    const int m = i / SEQ, s = i % SEQ;
    s_tok[wave][m][s] = (unsigned short)x[(size_t)(seq0 + m) * SEQ + s];
  }

  for (int d = 0; d < 2; ++d) {
    // stage this direction's xW table into LDS (cross-wave shared -> barriers)
    __syncthreads();
    for (int i = tid; i < VOCAB * G4; i += WPB * 32) s_xW[i] = ws_xw[d * VOCAB * G4 + i];
    __syncthreads();

    // resident Whh B-matrix tiles: v16h each (N=lane&15, element j = K in table order), 8 tiles
    v16h Bm[8];
    #pragma unroll
    for (int t = 0; t < 8; ++t)
      Bm[t] = ((const v16h*)ws_whh)[(d * 8 + t) * 32 + lane];

    v16h A;
    #pragma unroll
    for (int i = 0; i < 16; ++i) A[i] = (_Float16)0.0f;
    float c0[8], c1[8];
    #pragma unroll
    for (int r = 0; r < 8; ++r) { c0[r] = 0.0f; c1[r] = 0.0f; }

    for (int s = 0; s < SEQ; ++s) {
      const int step = d ? (SEQ - 1 - s) : s;

      // prefetch the 8 tokens for this step, then gather xW rows straight
      // into the WMMA C operands (D = A*B + C does the add for free)
      int toks[8];
      #pragma unroll
      for (int r = 0; r < 8; ++r) toks[r] = s_tok[wave][group * 8 + r][step];

      v8f acc[8];
      #pragma unroll
      for (int r = 0; r < 8; ++r) {
        const float* xw = &s_xW[toks[r] * G4];
        #pragma unroll
        for (int t = 0; t < 8; ++t) acc[t][r] = xw[t * 16 + nn];
      }
      #pragma unroll
      for (int t = 0; t < 8; ++t)
        acc[t] = __builtin_amdgcn_wmma_f32_16x16x32_f16(
            false, A, false, Bm[t], (short)0, acc[t], false, false);

      // elementwise: lane owns channels {nn, nn+16} for rows m = 8*group + r
      #pragma unroll
      for (int r = 0; r < 8; ++r) {
        const int m = group * 8 + r;
        c0[r] = sigmoid_t(acc[2][r]) * c0[r] + sigmoid_t(acc[0][r]) * tanh_approx(acc[4][r]);
        c1[r] = sigmoid_t(acc[3][r]) * c1[r] + sigmoid_t(acc[1][r]) * tanh_approx(acc[5][r]);
        const float h0 = sigmoid_t(acc[6][r]) * tanh_approx(c0[r]);
        const float h1 = sigmoid_t(acc[7][r]) * tanh_approx(c1[r]);
        s_hT[wave][m][nn]      = (_Float16)h0;
        s_hT[wave][m][16 + nn] = (_Float16)h1;
        if (s == SEQ - 1) {
          float* hout = d ? &s_hb[wave][m][0] : &s_hf[wave][m][0];
          hout[nn] = h0; hout[16 + nn] = h1;
        }
      }

      // s_hT is per-wave and DS ops within a wave are in-order; we only need
      // compiler ordering -> wavefront fences instead of block barriers.
      __builtin_amdgcn_fence(__ATOMIC_SEQ_CST, "wavefront");

      // reload A in the documented 16-bit A layout:
      // lane row M = nn; group 0 holds K {0..7,16..23}, group 1 K {8..15,24..31}
      const v8h alo = *(const v8h*)&s_hT[wave][nn][8 * group];
      const v8h ahi = *(const v8h*)&s_hT[wave][nn][16 + 8 * group];
      A = __builtin_shufflevector(alo, ahi, 0, 1, 2, 3, 4, 5, 6, 7,
                                  8, 9, 10, 11, 12, 13, 14, 15);
      __builtin_amdgcn_fence(__ATOMIC_SEQ_CST, "wavefront");
    }
  }
  __syncthreads();

  // ---- fused head: lanes 0..15 each handle one sequence -------------------
  if (lane < 16) {
    const int m = lane;

    unsigned short tk[SEQ];
    #pragma unroll
    for (int s = 0; s < SEQ; ++s) tk[s] = s_tok[wave][m][s];

    float total = 0.0f;
    #pragma unroll
    for (int s = 0; s < SEQ; ++s) total += (tk[s] != 0) ? 1.0f : 0.0f;

    const float inv_total = 1.0f / fmaxf(total, 1.0f);
    float ent = 0.0f, uniq = 0.0f;
    for (int v = 1; v < VOCAB; ++v) {
      int c = 0;
      #pragma unroll
      for (int s = 0; s < SEQ; ++s) c += (tk[s] == v) ? 1 : 0;
      if (c > 0) {
        uniq += 1.0f;
        const float p = (float)c * inv_total;
        ent -= p * __logf(p + 1e-8f);
      }
    }
    const float uniq_safe = fmaxf(fminf(uniq, (float)VOCAB), 1.0f);
    const float norm  = ent / (__logf(uniq_safe) + 1e-8f);
    const float featE = ((total > 1.0f) ? norm : 0.0f) * s_scal[0];
    const float featL = (total / 40.0f) * s_scal[1];

    float hid[32];
    #pragma unroll
    for (int j = 0; j < 32; ++j) hid[j] = s_b1[j];
    for (int k = 0; k < HID; ++k) {
      const float a  = s_hf[wave][m][k];
      const float bq = s_hb[wave][m][k];
      #pragma unroll
      for (int j = 0; j < 32; ++j)
        hid[j] += a * s_W1[j * 66 + k] + bq * s_W1[j * 66 + 32 + k];
    }
    #pragma unroll
    for (int j = 0; j < 32; ++j) {
      hid[j] += featE * s_W1[j * 66 + 64] + featL * s_W1[j * 66 + 65];
      hid[j] = fmaxf(hid[j], 0.0f);
    }

    float lg[4];
    #pragma unroll
    for (int e = 0; e < 4; ++e) {
      float sum = s_b2[e];
      #pragma unroll
      for (int j = 0; j < 32; ++j) sum += hid[j] * s_W2[e * 32 + j];
      lg[e] = sum;
    }
    const float mx = fmaxf(fmaxf(lg[0], lg[1]), fmaxf(lg[2], lg[3]));
    const float e0 = __expf(lg[0] - mx), e1 = __expf(lg[1] - mx);
    const float e2 = __expf(lg[2] - mx), e3 = __expf(lg[3] - mx);
    const float inv = 1.0f / (e0 + e1 + e2 + e3);
    float* op = out + (size_t)(seq0 + m) * 4;
    op[0] = e0 * inv; op[1] = e1 * inv; op[2] = e2 * inv; op[3] = e3 * inv;
  }
}

// ---------------------------------------------------------------------------
extern "C" void kernel_launch(void* const* d_in, const int* in_sizes, int n_in,
                              void* d_out, int out_size, void* d_ws, size_t ws_size,
                              hipStream_t stream)
{
  const int*   x    = (const int*)  d_in[0];
  const float* emb  = (const float*)d_in[1];
  const float* WihF = (const float*)d_in[2];
  const float* WhhF = (const float*)d_in[3];
  const float* bihF = (const float*)d_in[4];
  const float* bhhF = (const float*)d_in[5];
  const float* WihB = (const float*)d_in[6];
  const float* WhhB = (const float*)d_in[7];
  const float* bihB = (const float*)d_in[8];
  const float* bhhB = (const float*)d_in[9];
  const float* ew   = (const float*)d_in[10];
  const float* lw   = (const float*)d_in[11];
  const float* W1   = (const float*)d_in[12];
  const float* b1   = (const float*)d_in[13];
  const float* W2   = (const float*)d_in[14];
  const float* b2   = (const float*)d_in[15];
  float*       out  = (float*)d_out;

  float*    ws_xw  = (float*)d_ws;
  _Float16* ws_whh = (_Float16*)((char*)d_ws + WS_WHH_OFF);

  precompute_kernel<<<dim3(2 * VOCAB + 16), dim3(128), 0, stream>>>(
      emb, WihF, WhhF, bihF, bhhF, WihB, WhhB, bihB, bhhB, ws_xw, ws_whh);

  lstm_gate_kernel<<<dim3(BATCH / (WPB * SPW)), dim3(WPB * 32), 0, stream>>>(
      x, ws_xw, ws_whh, W1, b1, W2, b2, ew, lw, out);
}
